// DelayAttention_12128987644293
// MI455X (gfx1250) — matrix operand order repaired and checked by reading, not compiled, not run on hardware
//
#include <hip/hip_runtime.h>

typedef __attribute__((ext_vector_type(16))) __bf16 v16bf;
typedef __attribute__((ext_vector_type(8)))  __bf16 v8bf;
typedef __attribute__((ext_vector_type(8)))  float  v8f;
typedef __attribute__((ext_vector_type(4)))  float  v4f;

#define Bn  16
#define Tn  256
#define Nn  64
#define Dn  256
#define DKn 128
#define Sn  10
#define Pn  8

// ---------------- fragment helpers: every load is 2 x 16B ----------------

static __device__ __forceinline__ v16bf cat16(v8bf lo, v8bf hi) {
  return __builtin_shufflevector(lo, hi, 0, 1, 2, 3, 4, 5, 6, 7,
                                 8, 9, 10, 11, 12, 13, 14, 15);
}

// A-fragment (16x32) from row-major bf16 (row stride ldm, multiple of 8).
// lane<16: row=lane, K in {0..15}; lane>=16: row=lane-16, K {16..31}.
static __device__ __forceinline__ v16bf frag_rows(const __bf16* base, int ldm) {
  const int lane = threadIdx.x & 31;
  const __bf16* row = base + (lane & 15) * ldm + ((lane >> 4) << 4);
  return cat16(*(const v8bf*)row, *(const v8bf*)(row + 8));
}

// B-fragment (32x16) where source is n-major row-contiguous over k
// (element(k,n) = base[n*ldm + k]); used for Q*K^T.
static __device__ __forceinline__ v16bf frag_bT(const __bf16* base, int ldm) {
  const int lane = threadIdx.x & 31;
  const __bf16* row = base + (lane & 15) * ldm + ((lane >> 4) << 4);
  return cat16(*(const v8bf*)row, *(const v8bf*)(row + 8));
}

// Pre-swizzled fragment: 32 lanes x 16 contiguous bf16 (1 KB block).
static __device__ __forceinline__ v16bf frag_sw(const __bf16* base) {
  const int lane = threadIdx.x & 31;
  const __bf16* p = base + lane * 16;
  return cat16(*(const v8bf*)p, *(const v8bf*)(p + 8));
}

static __device__ __forceinline__ v8f wmma_bf16(v16bf a, v16bf b, v8f c) {
  return __builtin_amdgcn_wmma_f32_16x16x32_bf16(false, a, false, b, (short)0, c,
                                                 false, false);
}

// ---------------------------------------------------------------------------
// Kernel P: m = patterns @ Wm.T + bm ; c_sum = (sum_s patterns) @ Wc.T + S*bc
// ---------------------------------------------------------------------------
__global__ void __launch_bounds__(128) patterns_kernel(
    const float* __restrict__ patterns,
    const float* __restrict__ Wm, const float* __restrict__ bm,
    const float* __restrict__ Wc, const float* __restrict__ bc,
    __bf16* __restrict__ m_bf, __bf16* __restrict__ csum_bf) {
  const int r = blockIdx.x;
  const int j = threadIdx.x;
  if (r < Pn * Sn) {
    const int p = r / Sn, s = r % Sn;
    const float* pat = patterns + ((size_t)p * Sn + s) * Dn;
    const float* w   = Wm + (size_t)j * Dn;
    float acc = bm[j];
    for (int k = 0; k < Dn; ++k) acc += pat[k] * w[k];
    m_bf[(size_t)p * (Sn * DKn) + s * DKn + j] = (__bf16)acc;
  } else {
    const int p = r - Pn * Sn;
    const float* pat = patterns + (size_t)p * Sn * Dn;
    const float* w   = Wc + (size_t)j * Dn;
    float acc = (float)Sn * bc[j];
    for (int k = 0; k < Dn; ++k) {
      float ps = 0.f;
#pragma unroll
      for (int s = 0; s < Sn; ++s) ps += pat[s * Dn + k];
      acc += ps * w[k];
    }
    csum_bf[(size_t)p * DKn + j] = (__bf16)acc;
  }
}

// ---------------------------------------------------------------------------
// Swizzle m (as 1280x8 B-matrix, padded to 16 cols) and c_sum (as 32x128
// B-matrix, k padded 8->32) into WMMA fragment layout.
// m_sw: [ks=40][lane=32][16] ; csum_sw: [nt=8][lane=32][16]
// ---------------------------------------------------------------------------
__global__ void __launch_bounds__(256) mswz_kernel(
    const __bf16* __restrict__ m_bf, const __bf16* __restrict__ csum_bf,
    __bf16* __restrict__ m_sw, __bf16* __restrict__ csum_sw) {
  const int tid = blockIdx.x * 256 + threadIdx.x;
  const int MSW = 40 * 32 * 16;
  if (tid < MSW) {
    const int i = tid & 15, lane = (tid >> 4) & 31, ks = tid >> 9;
    const int n = lane & 15;
    const int k = ks * 32 + ((lane >> 4) << 4) + i;
    m_sw[tid] = (n < Pn) ? m_bf[(size_t)n * (Sn * DKn) + k] : (__bf16)0.f;
  } else if (tid < MSW + 8 * 32 * 16) {
    const int t2 = tid - MSW;
    const int i = t2 & 15, lane = (t2 >> 4) & 31, nt = t2 >> 9;
    const int n = nt * 16 + (lane & 15);
    const int k = ((lane >> 4) << 4) + i;
    csum_sw[t2] = (k < Pn) ? csum_bf[(size_t)k * DKn + n] : (__bf16)0.f;
  }
}

// ---------------------------------------------------------------------------
// Swizzle the 4 projection weight matrices into B-fragment layout:
// Wsw[ntile=32][ks=8][lane=32][16], ntile = global 16-col tile of the 512-wide
// fused output (Q|K|V|u), ks = 32-wide k step.  256 KB, L2-resident.
// ---------------------------------------------------------------------------
__global__ void __launch_bounds__(256) weights_kernel(
    const float* __restrict__ WQ, const float* __restrict__ WK,
    const float* __restrict__ WV, const float* __restrict__ Wu,
    __bf16* __restrict__ Wsw) {
  const int nt   = blockIdx.x;        // 0..31
  const int ks   = threadIdx.x >> 5;  // 0..7
  const int lane = threadIdx.x & 31;
  const int proj = nt >> 3;
  const float* W = proj == 0 ? WQ : proj == 1 ? WK : proj == 2 ? WV : Wu;
  const int col = (nt & 7) * 16 + (lane & 15);           // col within proj
  const int k0  = ks * 32 + ((lane >> 4) << 3);
  const float* src = W + (size_t)col * Dn + k0;          // k-contiguous
  __bf16* dst = Wsw + (((size_t)nt * 8 + ks) * 32 + lane) * 16;
  v8bf lo, hi;
#pragma unroll
  for (int i = 0; i < 8; ++i) lo[i] = (__bf16)src[i];
#pragma unroll
  for (int i = 0; i < 8; ++i) hi[i] = (__bf16)src[16 + i];
  *(v8bf*)dst = lo;
  *(v8bf*)(dst + 8) = hi;
}

// ---------------------------------------------------------------------------
// Kernel A: fused Q/K/V/u GEMM (262144 x 256) x (256 x 512). No LDS, no
// barriers: A-frags straight from x (fp32->bf16), B-frags from Wsw (L2).
// Block 256 thr = 8 waves; tile 128(M) x 64(N); wave owns 16 rows x 64 cols.
// V is written directly in B-fragment layout (Vsw) for the attention kernel.
// ---------------------------------------------------------------------------
__global__ void __launch_bounds__(256) proj_kernel(
    const float* __restrict__ x, const __bf16* __restrict__ Wsw,
    const float* __restrict__ bQ, const float* __restrict__ bK,
    const float* __restrict__ bV, const float* __restrict__ bu,
    __bf16* __restrict__ Qb, __bf16* __restrict__ Kb,
    __bf16* __restrict__ Vsw, __bf16* __restrict__ Ub) {
  const int tid  = threadIdx.x;
  const int wave = tid >> 5;
  const int lane = tid & 31;
  const int nb   = blockIdx.x;        // 0..7 (cols nb*64..+63 of 512)
  const size_t mb = blockIdx.y;       // 0..2047
  const int proj    = nb >> 1;
  const int colbase = (nb & 1) * 64;
  const float* bias = proj == 0 ? bQ : proj == 1 ? bK : proj == 2 ? bV : bu;

  const size_t rowTile = mb * 128 + wave * 16;
  const float* xrow = x + (rowTile + (lane & 15)) * (size_t)Dn + ((lane >> 4) << 3);
  const __bf16* wb  = Wsw + ((size_t)nb * 4) * 8 * 32 * 16;

  v8f c0 = {}, c1 = {}, c2 = {}, c3 = {};
#pragma unroll 1
  for (int kk = 0; kk < 8; ++kk) {
    const float* p = xrow + kk * 32;
    v4f f0 = *(const v4f*)(p);
    v4f f1 = *(const v4f*)(p + 4);
    v4f f2 = *(const v4f*)(p + 16);
    v4f f3 = *(const v4f*)(p + 20);
    v16bf a;
#pragma unroll
    for (int i = 0; i < 4; ++i) {
      a[i]      = (__bf16)f0[i];
      a[4 + i]  = (__bf16)f1[i];
      a[8 + i]  = (__bf16)f2[i];
      a[12 + i] = (__bf16)f3[i];
    }
    const __bf16* wk = wb + (size_t)kk * 32 * 16;
    c0 = wmma_bf16(a, frag_sw(wk + 0 * 8 * 32 * 16), c0);
    c1 = wmma_bf16(a, frag_sw(wk + 1 * 8 * 32 * 16), c1);
    c2 = wmma_bf16(a, frag_sw(wk + 2 * 8 * 32 * 16), c2);
    c3 = wmma_bf16(a, frag_sw(wk + 3 * 8 * 32 * 16), c3);
  }

  const int colL    = lane & 15;
  const int half    = (lane >> 4);
  const int rowBase = wave * 16 + half * 8;
  v8f cs[4] = {c0, c1, c2, c3};

  if (proj == 2) {
    // store V in B-fragment layout: Vsw[bt][kk2][nt2][lane'][i]
    const size_t bt  = mb * 2 + (wave >> 2);
    const int kk2    = (wave >> 1) & 1;     // head>=32 ?
    const int laneP  = colL + ((wave & 1) << 4);
    const int i0     = half * 8;
#pragma unroll
    for (int nt = 0; nt < 4; ++nt) {
      const int nt2 = (nb & 1) * 4 + nt;    // global 16-col d tile
      const float bv = bias[colbase + nt * 16 + colL];
      v8bf pk;
#pragma unroll
      for (int v = 0; v < 8; ++v) pk[v] = (__bf16)(cs[nt][v] + bv);
      *(v8bf*)(Vsw + ((((bt * 2 + kk2) * 8 + nt2) * 32 + laneP) * 16 + i0)) = pk;
    }
  } else {
    __bf16* dst = proj == 0 ? Qb : proj == 1 ? Kb : Ub;
#pragma unroll
    for (int nt = 0; nt < 4; ++nt) {
      const int col = colbase + nt * 16 + colL;
      const float bv = bias[col];
#pragma unroll
      for (int v = 0; v < 8; ++v)
        dst[(rowTile - wave * 16 + rowBase + v) * (size_t)DKn + col] =
            (__bf16)(cs[nt][v] + bv);
    }
  }
}

// ---------------------------------------------------------------------------
// Kernel C: per (b,t<246): sim = U_win(64x1280) @ m^T, softmax over P,
// r = w @ c_sum, K[b,t+S] += r.  Barrier-free WMMA loop; all fragment
// loads are wide; K RMW made contiguous through an LDS f32 buffer.
// ---------------------------------------------------------------------------
__global__ void __launch_bounds__(128) simr_kernel(
    const __bf16* __restrict__ Ub, const __bf16* __restrict__ m_sw,
    const __bf16* __restrict__ csum_sw, __bf16* __restrict__ Kb) {
  __shared__ float  simS[64][16];        // 4 KB
  __shared__ __bf16 Wsw2[4][32][16];     // softmax weights, frag layout, 4 KB
  __shared__ float  rS[64][128];         // r accumulator, 32 KB
  const int tid  = threadIdx.x;
  const int wave = tid >> 5;
  const int lane = tid & 31;
  const int t = blockIdx.x;              // 0..245
  const int b = blockIdx.y;              // 0..15
  const size_t bt = (size_t)b * Tn + t;

  v8f c = {};
#pragma unroll 1
  for (int ks = 0; ks < 40; ++ks) {      // contract S*DK = 1280
    const int s  = ks >> 2;
    const int d0 = (ks & 3) * 32;
    v16bf a  = frag_rows(Ub + ((bt + s) * Nn + wave * 16) * (size_t)DKn + d0, DKn);
    v16bf bb = frag_sw(m_sw + (size_t)ks * 32 * 16);
    c = wmma_bf16(a, bb, c);
  }

  const int colL    = lane & 15;
  const int rowBase = wave * 16 + ((lane >> 4) << 3);
#pragma unroll
  for (int v = 0; v < 8; ++v) simS[rowBase + v][colL] = c[v];

  {  // zero-fill the swizzled softmax-weight fragments (padding lanes/slots)
    __bf16* flat = &Wsw2[0][0][0];
    v8bf z = {};
    *(v8bf*)(flat + tid * 16) = z;
    *(v8bf*)(flat + tid * 16 + 8) = z;
  }
  __syncthreads();

  if (tid < Nn) {                        // softmax over P per head row
    float mx = -1e30f;
#pragma unroll
    for (int p = 0; p < Pn; ++p) mx = fmaxf(mx, simS[tid][p]);
    float e[Pn]; float sum = 0.f;
#pragma unroll
    for (int p = 0; p < Pn; ++p) { e[p] = __expf(simS[tid][p] - mx); sum += e[p]; }
    const float inv = 1.f / sum;
    v8bf pk;
#pragma unroll
    for (int p = 0; p < Pn; ++p) pk[p] = (__bf16)(e[p] * inv);
    // A-frag slot for k=0..7 of row tid: tile=tid>>4, lane'=tid&15, i=0..7
    *(v8bf*)&Wsw2[tid >> 4][tid & 15][0] = pk;
  }
  __syncthreads();

  v16bf a2 = frag_sw(&Wsw2[wave][0][0]);
#pragma unroll
  for (int nt = 0; nt < 8; ++nt) {       // r = w @ c_sum (128 cols)
    v8f rc = {};
    rc = wmma_bf16(a2, frag_sw(csum_sw + (size_t)nt * 32 * 16), rc);
#pragma unroll
    for (int v = 0; v < 8; ++v) rS[rowBase + v][nt * 16 + colL] = rc[v];
  }
  __syncthreads();

  // contiguous bf16 RMW of K[b, t+S]: thread owns half a head row
  const int row = tid >> 1;
  const int c0  = (tid & 1) * 64;
  const size_t kbase = ((bt + Sn) * Nn + row) * (size_t)DKn + c0;
#pragma unroll
  for (int g = 0; g < 8; ++g) {
    v8bf kv = *(const v8bf*)(Kb + kbase + g * 8);
    v8bf nv;
#pragma unroll
    for (int i = 0; i < 8; ++i)
      nv[i] = (__bf16)((float)kv[i] + rS[row][c0 + g * 8 + i]);
    *(v8bf*)(Kb + kbase + g * 8) = nv;
  }
}

// ---------------------------------------------------------------------------
// Kernel D: per (b,t): scores = Q K^T / sqrt(128), softmax over m,
// out = attn @ V.  Q/K fragments straight from global (row-contiguous);
// attn written pre-swizzled to LDS; V fragments from Vsw. LDS = 24 KB.
// ---------------------------------------------------------------------------
__global__ void __launch_bounds__(128) attn_kernel(
    const __bf16* __restrict__ Qb, const __bf16* __restrict__ Kb,
    const __bf16* __restrict__ Vsw, float* __restrict__ out) {
  __shared__ float  Ss[64][64];          // scores, 16 KB
  __shared__ __bf16 Asw[4][2][32][16];   // attn A-frags, 8 KB
  const int tid  = threadIdx.x;
  const int wave = tid >> 5;
  const int lane = tid & 31;
  const size_t bt = blockIdx.x;
  const size_t base = bt * (size_t)(Nn * DKn);
  const __bf16* Qbase = Qb + base;
  const __bf16* Kbase = Kb + base;

  v8f sc[4] = {};
#pragma unroll
  for (int kk = 0; kk < 4; ++kk) {
    v16bf a = frag_rows(Qbase + (size_t)(wave * 16) * DKn + kk * 32, DKn);
#pragma unroll
    for (int nt = 0; nt < 4; ++nt)
      sc[nt] = wmma_bf16(a, frag_bT(Kbase + (size_t)(nt * 16) * DKn + kk * 32, DKn),
                         sc[nt]);
  }

  const int colL    = lane & 15;
  const int rowBase = wave * 16 + ((lane >> 4) << 3);
  const float scale = 0.08838834764831845f;   // 1/sqrt(128)
#pragma unroll
  for (int nt = 0; nt < 4; ++nt)
#pragma unroll
    for (int v = 0; v < 8; ++v)
      Ss[rowBase + v][nt * 16 + colL] = sc[nt][v] * scale;
  __syncthreads();

  if (tid < Nn) {                        // softmax over the 64 key-heads
    const int n = tid;
    float mx = -1e30f;
    for (int m = 0; m < Nn; ++m) mx = fmaxf(mx, Ss[n][m]);
    float sum = 0.f;
    for (int m = 0; m < Nn; ++m) sum += __expf(Ss[n][m] - mx);
    const float inv = 1.f / sum;
#pragma unroll
    for (int c8 = 0; c8 < 8; ++c8) {     // write 8 probs per wide store,
      v8bf pk;                           // directly in A-frag layout
#pragma unroll
      for (int j = 0; j < 8; ++j)
        pk[j] = (__bf16)(__expf(Ss[n][c8 * 8 + j] - mx) * inv);
      *(v8bf*)&Asw[n >> 4][c8 >> 2][(n & 15) + ((c8 & 2) << 3)][(c8 & 1) << 3] = pk;
    }
  }
  __syncthreads();

  v8f oc[8] = {};
#pragma unroll
  for (int kk = 0; kk < 2; ++kk) {
    v16bf a = frag_sw(&Asw[wave][kk][0][0]);
#pragma unroll
    for (int nt = 0; nt < 8; ++nt)
      oc[nt] = wmma_bf16(a, frag_sw(Vsw + (((bt * 2 + kk) * 8 + nt) * 32) * 16),
                         oc[nt]);
  }
#pragma unroll
  for (int nt = 0; nt < 8; ++nt)
#pragma unroll
    for (int v = 0; v < 8; ++v)
      out[base + (size_t)(rowBase + v) * DKn + nt * 16 + colL] = oc[nt][v];
}

// ---------------------------------------------------------------------------
extern "C" void kernel_launch(void* const* d_in, const int* in_sizes, int n_in,
                              void* d_out, int out_size, void* d_ws, size_t ws_size,
                              hipStream_t stream) {
  (void)in_sizes; (void)n_in; (void)out_size; (void)ws_size;
  const float* x  = (const float*)d_in[0];
  const float* WQ = (const float*)d_in[1];
  const float* bQ = (const float*)d_in[2];
  const float* WK = (const float*)d_in[3];
  const float* bK = (const float*)d_in[4];
  const float* WV = (const float*)d_in[5];
  const float* bV = (const float*)d_in[6];
  const float* Wu = (const float*)d_in[7];
  const float* bu = (const float*)d_in[8];
  const float* Wm = (const float*)d_in[9];
  const float* bm = (const float*)d_in[10];
  const float* Wc = (const float*)d_in[11];
  const float* bc = (const float*)d_in[12];
  const float* patterns = (const float*)d_in[13];
  float* out = (float*)d_out;

  const size_t M = (size_t)Bn * Tn * Nn;                 // 262144 rows
  const size_t projBytes = M * DKn * sizeof(__bf16);     // 64 MB each
  unsigned char* ws = (unsigned char*)d_ws;
  __bf16* Qb   = (__bf16*)(ws);
  __bf16* Kb   = (__bf16*)(ws + 1 * projBytes);
  __bf16* Vsw  = (__bf16*)(ws + 2 * projBytes);
  __bf16* Ub   = (__bf16*)(ws + 3 * projBytes);
  unsigned char* tail = ws + 4 * projBytes;
  __bf16* m_bf    = (__bf16*)(tail);                     // 20480 B
  __bf16* csum_bf = (__bf16*)(tail + 20480);             //  2048 B
  __bf16* m_sw    = (__bf16*)(tail + 22528);             // 40960 B
  __bf16* csum_sw = (__bf16*)(tail + 63488);             //  8192 B
  __bf16* Wsw     = (__bf16*)(tail + 71680);             // 262144 B

  patterns_kernel<<<dim3(Pn * Sn + Pn), dim3(128), 0, stream>>>(
      patterns, Wm, bm, Wc, bc, m_bf, csum_bf);
  mswz_kernel<<<dim3(96), dim3(256), 0, stream>>>(m_bf, csum_bf, m_sw, csum_sw);
  weights_kernel<<<dim3(32), dim3(256), 0, stream>>>(WQ, WK, WV, Wu, Wsw);
  proj_kernel<<<dim3(8, (unsigned)(M / 128)), dim3(256), 0, stream>>>(
      x, Wsw, bQ, bK, bV, bu, Qb, Kb, Vsw, Ub);
  simr_kernel<<<dim3(Tn - Sn, Bn), dim3(128), 0, stream>>>(Ub, m_sw, csum_sw, Kb);
  attn_kernel<<<dim3(Bn * Tn), dim3(128), 0, stream>>>(Qb, Kb, Vsw, out);
}
